// RNNClassifier_82282983457014
// MI455X (gfx1250) — compile-verified, matrix operand order checked by reading, not run on hardware
//
#include <hip/hip_runtime.h>
#include <hip/hip_bf16.h>

// Problem sizes (fixed by the reference).
#define BDIM 2048   // batch
#define FDIM 2048   // feature
#define K2   4096   // 2F (concat dim)
#define CDIM 50257  // classes

typedef __attribute__((ext_vector_type(16))) __bf16 v16bf;
typedef __attribute__((ext_vector_type(8)))  __bf16 v8bf;
typedef __attribute__((ext_vector_type(2)))  __bf16 v2bf;
typedef __attribute__((ext_vector_type(8)))  float  v8f;

// Scheduling fence: keep global-load issue / WMMA block / LDS staging phases
// from being interleaved by the post-RA scheduler.
__device__ __forceinline__ void sched_fence() {
#if __has_builtin(__builtin_amdgcn_sched_barrier)
  __builtin_amdgcn_sched_barrier(0);
#endif
}

// fp32 -> bf16 with round-to-nearest-even (scalar path, epilogue only).
__device__ __forceinline__ unsigned short f2bf(float f) {
  unsigned u = __builtin_bit_cast(unsigned, f);
  u += 0x7FFFu + ((u >> 16) & 1u);
  return (unsigned short)(u >> 16);
}

// Packed fp32x2 -> bf16x2 (RNE).  gfx1250 lacks v_cvt_pk_bf16_f32 (probed via
// __has_builtin in round 3), so round both words and merge their high halves
// with a single v_perm_b32: result = {r0[23:16],r0[31:24],r1[23:16],r1[31:24]}.
__device__ __forceinline__ unsigned pk2bf(float x, float y) {
#if __has_builtin(__builtin_amdgcn_cvt_pk_bf16_f32)
  v2bf p = __builtin_amdgcn_cvt_pk_bf16_f32(x, y);
  return __builtin_bit_cast(unsigned, p);
#else
  unsigned u0 = __builtin_bit_cast(unsigned, x);
  unsigned u1 = __builtin_bit_cast(unsigned, y);
  unsigned r0 = u0 + 0x7FFFu + ((u0 >> 16) & 1u);
  unsigned r1 = u1 + 0x7FFFu + ((u1 >> 16) & 1u);
  // byte sel: {r0.b2, r0.b3, r1.b2, r1.b3}  (src0 = r1 -> bytes 4..7)
  return __builtin_amdgcn_perm(r1, r0, 0x07060302u);
#endif
}

// Assemble a 16-element bf16 WMMA fragment from two contiguous 16-byte LDS
// chunks (ISA 7.12.2: 16-bit A 16x32 layout — VGPR0..3 hold K=ko..ko+7,
// VGPR4..7 hold K=ko+16..ko+23, ko = 8*(lane>>4); B mirrored with N<->M,
// staged transposed [n][k] in LDS so both fragments are two ds_load_b128).
__device__ __forceinline__ v16bf frag16(const unsigned short* p0,
                                        const unsigned short* p1) {
  v8bf lo = *(const v8bf*)p0;
  v8bf hi = *(const v8bf*)p1;
  return __builtin_shufflevector(lo, hi, 0,1,2,3,4,5,6,7,8,9,10,11,12,13,14,15);
}

__device__ __forceinline__ float fsigmoid(float x) {
  return 1.0f / (1.0f + __expf(-x));
}
__device__ __forceinline__ float ftanh(float x) {
  return 1.0f - 2.0f / (__expf(2.0f * x) + 1.0f);
}

// ---------------------------------------------------------------------------
// K1: xh[b] = concat(emb[X[b]], h[b]) as bf16   ([BDIM][K2])
// ---------------------------------------------------------------------------
__global__ __launch_bounds__(256) void k_build_xh(
    const int* __restrict__ X, const float* __restrict__ h,
    const float* __restrict__ emb, unsigned short* __restrict__ xh) {
  const int b = blockIdx.x;
  const float4* s0 = (const float4*)(emb + (long)X[b] * FDIM);
  const float4* s1 = (const float4*)(h   + (long)b * FDIM);
  ushort4* dst = (ushort4*)(xh + (long)b * K2);
  for (int i = threadIdx.x; i < FDIM / 4; i += 256) {
    float4 v = s0[i];
    uint2 pv = make_uint2(pk2bf(v.x, v.y), pk2bf(v.z, v.w));
    dst[i] = __builtin_bit_cast(ushort4, pv);
    float4 w = s1[i];
    uint2 pw = make_uint2(pk2bf(w.x, w.y), pk2bf(w.z, w.w));
    dst[i + FDIM / 4] = __builtin_bit_cast(ushort4, pw);
  }
}

// ---------------------------------------------------------------------------
// K2: fused 4-gate GEMM + LSTM pointwise, double-buffered LDS.
// Block tile 128(M) x 64(N); 8 waves = 4(M) x 2(N); wave tile 32x32 x 4 gates
// = 16 WMMA per K-step.  acc: 4 gates x 2 mi x 2 jj = 128 VGPRs.
// ---------------------------------------------------------------------------
__global__ __launch_bounds__(256) void k_lstm_gates(
    const unsigned short* __restrict__ xh,
    const float* __restrict__ Wg0, const float* __restrict__ Wg1,
    const float* __restrict__ Wg2, const float* __restrict__ Wg3,
    const float* __restrict__ bg0, const float* __restrict__ bg1,
    const float* __restrict__ bg2, const float* __restrict__ bg3,
    const float* __restrict__ c_in,
    float* __restrict__ h_out, float* __restrict__ c_out,
    unsigned short* __restrict__ h_bf) {
  __shared__ __align__(16) unsigned short lA[2][128][32];     // [buf][m][k]
  __shared__ __align__(16) unsigned short lB[2][4][64][32];   // [buf][g][n][k]

  const int t  = threadIdx.x;
  const int m0 = blockIdx.x * 128;  // M fast in grid -> W panel reused via L2
  const int n0 = blockIdx.y * 64;
  const float* W[4] = {Wg0, Wg1, Wg2, Wg3};

  const int lane = t & 31;
  const int wv   = t >> 5;
  const int wM   = (wv & 3) * 32;   // 4 waves along M, 32 rows each
  const int wN   = (wv >> 2) * 32;  // 2 waves along N, 32 cols each
  const int fr   = lane & 15;
  const int ko   = (lane >> 4) * 8;

  // Staging registers for the software pipeline (global -> regs -> LDS).
  ushort4 ra[4];
  float4  rb[4][2];

  auto ldregs = [&](int k0) {
#pragma unroll
    for (int i = 0; i < 4; ++i) {
      int j = t + 256 * i;
      int m = j >> 3, kc = (j & 7) * 4;
      ra[i] = *(const ushort4*)&xh[(long)(m0 + m) * K2 + k0 + kc];
    }
#pragma unroll
    for (int g = 0; g < 4; ++g)
#pragma unroll
      for (int i = 0; i < 2; ++i) {
        int j = t + 256 * i;
        int kk = j >> 4, nn = (j & 15) * 4;
        rb[g][i] = *(const float4*)&W[g][(long)(k0 + kk) * FDIM + n0 + nn];
      }
  };
  auto stlds = [&](int buf) {
#pragma unroll
    for (int i = 0; i < 4; ++i) {
      int j = t + 256 * i;
      int m = j >> 3, kc = (j & 7) * 4;
      *(ushort4*)&lA[buf][m][kc] = ra[i];
    }
#pragma unroll
    for (int g = 0; g < 4; ++g)
#pragma unroll
      for (int i = 0; i < 2; ++i) {
        int j = t + 256 * i;
        int kk = j >> 4, nn = (j & 15) * 4;
        unsigned p01 = pk2bf(rb[g][i].x, rb[g][i].y);
        unsigned p23 = pk2bf(rb[g][i].z, rb[g][i].w);
        lB[buf][g][nn + 0][kk] = (unsigned short)p01;
        lB[buf][g][nn + 1][kk] = (unsigned short)(p01 >> 16);
        lB[buf][g][nn + 2][kk] = (unsigned short)p23;
        lB[buf][g][nn + 3][kk] = (unsigned short)(p23 >> 16);
      }
  };

  const v8f vzero = {0.f, 0.f, 0.f, 0.f, 0.f, 0.f, 0.f, 0.f};
  v8f acc[4][2][2];
#pragma unroll
  for (int g = 0; g < 4; ++g)
#pragma unroll
    for (int mi = 0; mi < 2; ++mi)
#pragma unroll
      for (int jj = 0; jj < 2; ++jj) acc[g][mi][jj] = vzero;

  ldregs(0);
  stlds(0);
  __syncthreads();

  const int NK = K2 / 32;
  for (int kt = 0; kt < NK; ++kt) {
    const int cur = kt & 1;
    // Phase 1: issue next tile's global loads (latency hidden by WMMA block).
    if (kt + 1 < NK) ldregs((kt + 1) * 32);
    if (kt + 2 < NK) {
#pragma unroll
      for (int g = 0; g < 4; ++g)
        __builtin_prefetch(
            &W[g][(long)((kt + 2) * 32 + (t >> 3)) * FDIM + n0 + (t & 7) * 8],
            0, 0);
    }
    sched_fence();  // loads stay above the matrix block

    // Phase 2: A fragments once, then 4 WMMAs back-to-back per gate (2 per B fragment).
    v16bf af[2];
#pragma unroll
    for (int mi = 0; mi < 2; ++mi)
      af[mi] = frag16(&lA[cur][wM + mi * 16 + fr][ko],
                      &lA[cur][wM + mi * 16 + fr][ko + 16]);
#pragma unroll
    for (int g = 0; g < 4; ++g) {
      v16bf b0 = frag16(&lB[cur][g][wN + fr][ko],
                        &lB[cur][g][wN + fr][ko + 16]);
      v16bf b1 = frag16(&lB[cur][g][wN + 16 + fr][ko],
                        &lB[cur][g][wN + 16 + fr][ko + 16]);
#pragma unroll
      for (int mi = 0; mi < 2; ++mi) {
        acc[g][mi][0] = __builtin_amdgcn_wmma_f32_16x16x32_bf16(
            false, af[mi], false, b0, (short)0, acc[g][mi][0], false, false);
        acc[g][mi][1] = __builtin_amdgcn_wmma_f32_16x16x32_bf16(
            false, af[mi], false, b1, (short)0, acc[g][mi][1], false, false);
      }
    }
    sched_fence();  // conversions/stores stay below the matrix block

    // Phase 3: convert + store to the other buffer (last read before the
    // previous barrier, so no race).  One barrier per K-step.
    if (kt + 1 < NK) stlds(cur ^ 1);
    __syncthreads();
  }

  // Epilogue: bias + LSTM pointwise. C/D layout: VGPR v of lane L holds
  // (row = v + 8*(L>>4), col = L&15).
  const int hi = lane >> 4;
#pragma unroll
  for (int jj = 0; jj < 2; ++jj) {
    const int n = n0 + wN + jj * 16 + fr;
    const float bF = bg0[n], bI = bg1[n], bO = bg2[n], bG = bg3[n];
#pragma unroll
    for (int mi = 0; mi < 2; ++mi)
#pragma unroll
      for (int v = 0; v < 8; ++v) {
        const int m   = m0 + wM + mi * 16 + v + 8 * hi;
        const long ix = (long)m * FDIM + n;
        float fg = fsigmoid(acc[0][mi][jj][v] + bF);
        float ig = fsigmoid(acc[1][mi][jj][v] + bI);
        float og = fsigmoid(acc[2][mi][jj][v] + bO);
        float gg = ftanh(acc[3][mi][jj][v] + bG);
        float cn = fg * c_in[ix] + ig * gg;
        float hn = og * ftanh(cn);
        c_out[ix] = cn;
        h_out[ix] = hn;
        h_bf[ix]  = f2bf(hn);
      }
  }
}

// ---------------------------------------------------------------------------
// K3: y = h_new(bf16) @ Wcls + bcls, double-buffered LDS.
// Block tile 128(M) x 256(N); 8 waves = 2(M) x 4(N); wave tile 64x64
// = 16 WMMA per K-step (4 back-to-back per B fragment).
// ---------------------------------------------------------------------------
__global__ __launch_bounds__(256) void k_cls_gemm(
    const unsigned short* __restrict__ A,  // [BDIM][FDIM] bf16
    const float* __restrict__ W,           // [FDIM][CDIM] fp32
    const float* __restrict__ bias, float* __restrict__ Y) {
  __shared__ __align__(16) unsigned short lA[2][128][32];  // [buf][m][k]
  __shared__ __align__(16) unsigned short lB[2][256][32];  // [buf][n][k]

  const int t  = threadIdx.x;
  const int m0 = blockIdx.x * 128;  // M fast -> W panel (2 MB) reused via L2
  const int n0 = blockIdx.y * 256;
  const int lane = t & 31, wv = t >> 5;
  const int wM = (wv & 1) * 64;     // 2 waves along M, 64 rows each
  const int wN = (wv >> 1) * 64;    // 4 waves along N, 64 cols each
  const int fr = lane & 15, ko = (lane >> 4) * 8;

  // Only the last N-panel (1 of 197) needs the tail guard; specialize on the
  // uniform condition so hot blocks stage with no per-element v_cmp/cndmask.
  const bool edge = (n0 + 256 > CDIM);

  ushort4 ra[4];
  float   rbv[8][4];

  auto ldregs = [&](int k0) {
#pragma unroll
    for (int i = 0; i < 4; ++i) {
      int j = t + 256 * i;
      int m = j >> 3, kc = (j & 7) * 4;
      ra[i] = *(const ushort4*)&A[(long)(m0 + m) * FDIM + k0 + kc];
    }
    // W rows have odd length (50257): scalar dword loads.
    if (!edge) {
#pragma unroll
      for (int i = 0; i < 8; ++i) {
        int j  = t + 256 * i;
        int kk = j >> 6, nn = (j & 63) * 4;
        const float* rp = &W[(long)(k0 + kk) * CDIM + n0 + nn];
#pragma unroll
        for (int e = 0; e < 4; ++e) rbv[i][e] = rp[e];
      }
    } else {
#pragma unroll
      for (int i = 0; i < 8; ++i) {
        int j  = t + 256 * i;
        int kk = j >> 6, nn = (j & 63) * 4;
        const long rbase = (long)(k0 + kk) * CDIM;
#pragma unroll
        for (int e = 0; e < 4; ++e) {
          int n = n0 + nn + e;
          rbv[i][e] = (n < CDIM) ? W[rbase + n] : 0.0f;
        }
      }
    }
  };
  auto stlds = [&](int buf) {
#pragma unroll
    for (int i = 0; i < 4; ++i) {
      int j = t + 256 * i;
      int m = j >> 3, kc = (j & 7) * 4;
      *(ushort4*)&lA[buf][m][kc] = ra[i];
    }
#pragma unroll
    for (int i = 0; i < 8; ++i) {
      int j  = t + 256 * i;
      int kk = j >> 6, nn = (j & 63) * 4;
      unsigned p01 = pk2bf(rbv[i][0], rbv[i][1]);
      unsigned p23 = pk2bf(rbv[i][2], rbv[i][3]);
      lB[buf][nn + 0][kk] = (unsigned short)p01;
      lB[buf][nn + 1][kk] = (unsigned short)(p01 >> 16);
      lB[buf][nn + 2][kk] = (unsigned short)p23;
      lB[buf][nn + 3][kk] = (unsigned short)(p23 >> 16);
    }
  };

  const v8f vzero = {0.f, 0.f, 0.f, 0.f, 0.f, 0.f, 0.f, 0.f};
  v8f acc[4][4];
#pragma unroll
  for (int mi = 0; mi < 4; ++mi)
#pragma unroll
    for (int jj = 0; jj < 4; ++jj) acc[mi][jj] = vzero;

  ldregs(0);
  stlds(0);
  __syncthreads();

  const int NK = FDIM / 32;
  for (int kt = 0; kt < NK; ++kt) {
    const int cur = kt & 1;
    // Phase 1: next-tile global loads.
    if (kt + 1 < NK) ldregs((kt + 1) * 32);
    if (kt + 2 < NK) {
      int pn = n0 + (t & 7) * 32;
      if (pn < CDIM)
        __builtin_prefetch(&W[(long)((kt + 2) * 32 + (t >> 3)) * CDIM + pn],
                           0, 0);
    }
    sched_fence();

    // Phase 2: matrix block.
    v16bf af[4];
#pragma unroll
    for (int mi = 0; mi < 4; ++mi)
      af[mi] = frag16(&lA[cur][wM + mi * 16 + fr][ko],
                      &lA[cur][wM + mi * 16 + fr][ko + 16]);
#pragma unroll
    for (int jj = 0; jj < 4; ++jj) {
      v16bf b = frag16(&lB[cur][wN + jj * 16 + fr][ko],
                       &lB[cur][wN + jj * 16 + fr][ko + 16]);
#pragma unroll
      for (int mi = 0; mi < 4; ++mi)
        acc[mi][jj] = __builtin_amdgcn_wmma_f32_16x16x32_bf16(
            false, af[mi], false, b, (short)0, acc[mi][jj], false, false);
    }
    sched_fence();

    // Phase 3: convert + stage into the other buffer.
    if (kt + 1 < NK) stlds(cur ^ 1);
    __syncthreads();
  }

  const int hi = lane >> 4;
#pragma unroll
  for (int jj = 0; jj < 4; ++jj) {
    const int n = n0 + wN + jj * 16 + fr;
    if (n < CDIM) {
      const float bv = bias[n];
#pragma unroll
      for (int mi = 0; mi < 4; ++mi)
#pragma unroll
        for (int v = 0; v < 8; ++v) {
          const int m = m0 + wM + mi * 16 + v + 8 * hi;
          Y[(long)m * CDIM + n] = acc[mi][jj][v] + bv;
        }
    }
  }
}

// ---------------------------------------------------------------------------
extern "C" void kernel_launch(void* const* d_in, const int* in_sizes, int n_in,
                              void* d_out, int out_size, void* d_ws,
                              size_t ws_size, hipStream_t stream) {
  (void)in_sizes; (void)n_in; (void)out_size; (void)ws_size;
  const int*   X    = (const int*)  d_in[0];
  const float* h    = (const float*)d_in[1];
  const float* c    = (const float*)d_in[2];
  const float* emb  = (const float*)d_in[3];
  const float* Wf   = (const float*)d_in[4];
  const float* bf_  = (const float*)d_in[5];
  const float* Wi   = (const float*)d_in[6];
  const float* bi_  = (const float*)d_in[7];
  const float* Wig  = (const float*)d_in[8];
  const float* big_ = (const float*)d_in[9];
  const float* Wog  = (const float*)d_in[10];
  const float* bog_ = (const float*)d_in[11];
  const float* Wcls = (const float*)d_in[12];
  const float* bcls = (const float*)d_in[13];

  // d_out = (y, h_new, c_new) flat.
  float* y     = (float*)d_out;
  float* h_out = y + (size_t)BDIM * CDIM;
  float* c_out = h_out + (size_t)BDIM * FDIM;

  // Workspace: xh bf16 (16 MB) + h_new bf16 (8 MB) = 24 MB.
  unsigned short* xh   = (unsigned short*)d_ws;
  unsigned short* h_bf = xh + (size_t)BDIM * K2;

  k_build_xh<<<BDIM, 256, 0, stream>>>(X, h, emb, xh);
  // Gate order per reference: f<-(Wf,bf), i<-(Wig,big), o<-(Wog,bog), g<-(Wi,bi)
  k_lstm_gates<<<dim3(BDIM / 128, FDIM / 64), 256, 0, stream>>>(
      xh, Wf, Wig, Wog, Wi, bf_, big_, bog_, bi_, c, h_out, c_out, h_bf);
  k_cls_gemm<<<dim3(BDIM / 128, (CDIM + 255) / 256), 256, 0, stream>>>(
      h_bf, Wcls, bcls, y);
}
